// MultiHeadAttention_20787641713121
// MI455X (gfx1250) — matrix operand order repaired and checked by reading, not compiled, not run on hardware
//
#include <hip/hip_runtime.h>

// MHA block: B=8, S=1024, D=512, H=8, DK=512 (d_model quirk).
// All GEMMs via v_wmma_f32_16x16x32_bf16 (fp32->bf16, fp32 accum).
// Attention: flash-style, O accum (16x512 f32) in VGPRs, K/V double-buffered
// into LDS with CDNA5 async global->LDS copies (ASYNCcnt), Q staged in LDS.

typedef __attribute__((ext_vector_type(16))) __bf16 bf16x16;
typedef __attribute__((ext_vector_type(8)))  __bf16 bf16x8;
typedef __attribute__((ext_vector_type(8)))  float  f32x8;

__device__ __forceinline__ f32x8 wmma_bf16(bf16x16 a, bf16x16 b, f32x8 c) {
  return __builtin_amdgcn_wmma_f32_16x16x32_bf16(false, a, false, b, (short)0, c, false, false);
}

// A operand (16x32, MxK): lane L -> row L&15, k chunks [(L>>4)*8, +8) and [+16, +24)
__device__ __forceinline__ bf16x16 ldA_bf16(const __bf16* tile, int ld, int lane) {
  const int r = lane & 15, kh = (lane >> 4) << 3;
  const __bf16* p = tile + (size_t)r * ld + kh;
  bf16x8 lo = *(const bf16x8*)(p);
  bf16x8 hi = *(const bf16x8*)(p + 16);
  return __builtin_shufflevector(lo, hi, 0,1,2,3,4,5,6,7,8,9,10,11,12,13,14,15);
}

// B operand (32x16, KxN): lane L -> col L&15, k chunks [(L>>4)*8, +8) and [+16, +24)
__device__ __forceinline__ bf16x16 ldB_bf16(const __bf16* tile, int ld, int lane) {
  const int n = lane & 15, kh = (lane >> 4) << 3;
  const __bf16* p = tile + (size_t)n * ld + kh;
  bf16x8 lo = *(const bf16x8*)(p);
  bf16x8 hi = *(const bf16x8*)(p + 16);
  return __builtin_shufflevector(lo, hi, 0,1,2,3,4,5,6,7,8,9,10,11,12,13,14,15);
}

__device__ __forceinline__ bf16x16 ldA_f32(const float* tile, int ld, int lane) {
  const int r = lane & 15, kh = (lane >> 4) << 3;
  const float* p = tile + (size_t)r * ld + kh;
  bf16x16 a;
#pragma unroll
  for (int i = 0; i < 8; ++i) { a[i] = (__bf16)p[i]; a[8 + i] = (__bf16)p[16 + i]; }
  return a;
}

__device__ __forceinline__ bf16x16 ldB_f32(const float* tile, int ld, int lane) {
  const int n = lane & 15, kh = (lane >> 4) << 3;
  const float* p = tile + (size_t)n * ld + kh;
  bf16x16 b;
#pragma unroll
  for (int i = 0; i < 8; ++i) { b[i] = (__bf16)p[i]; b[8 + i] = (__bf16)p[16 + i]; }
  return b;
}

// C/D 16x16 f32 tile layout: value(m,n): vgpr = m&7, lane = n + 16*(m>=8).
// Store as bf16 row-major via per-wave LDS transpose (256-elem region).
__device__ __forceinline__ void store_tile_bf16(f32x8 c, __bf16* out, int ld,
                                                __bf16* lds, int lane) {
  const int n = lane & 15, mh = (lane >> 4) << 3;
#pragma unroll
  for (int v = 0; v < 8; ++v) lds[(mh + v) * 16 + n] = (__bf16)c[v];
  asm volatile("s_wait_dscnt 0" ::: "memory");   // cross-lane LDS RAW (same wave)
  const int r = lane >> 1, cb = (lane & 1) << 3;
  bf16x8 row = *(const bf16x8*)(lds + r * 16 + cb);
  *(bf16x8*)(out + (size_t)r * ld + cb) = row;
  asm volatile("s_wait_dscnt 0" ::: "memory");   // WAR before region reuse
}

// CDNA5 async global->LDS copy, 16B per lane, tracked by ASYNCcnt.
__device__ __forceinline__ void async_cp16(unsigned ldsoff, const __bf16* g) {
  asm volatile("global_load_async_to_lds_b128 %0, %1, off"
               :: "v"(ldsoff), "v"(g) : "memory");
}
__device__ __forceinline__ unsigned lds_off(const void* p) {
  return (unsigned)(uintptr_t)p;   // LDS aperture: flat addr[31:0] == LDS offset
}

// ---------------- Projection GEMM: (8192x512 f32) @ Wt -> bf16 (8192x4096) ----------------
// transposed==1 stores per-head V^T: vt[((b*8+h)*512+dk)*1024 + s]  (free from C layout)
__global__ __launch_bounds__(256) void proj_kernel(const float* __restrict__ X,
                                                   const float* __restrict__ W,
                                                   __bf16* __restrict__ out,
                                                   int transposed) {
  __shared__ __align__(32) __bf16 lds[8 * 256];
  const int lane = threadIdx.x & 31, widx = threadIdx.x >> 5;
  const int wg = blockIdx.x * 8 + widx;
  const int mt = wg >> 6;   // 512 row tiles of 16 over 8192
  const int ng = wg & 63;   // 64 col groups of 64 over 4096
  const float* Xt = X + (size_t)mt * 16 * 512;
  f32x8 acc[4] = {};
#pragma unroll 4
  for (int kc = 0; kc < 512; kc += 32) {
    bf16x16 a = ldA_f32(Xt + kc, 512, lane);
#pragma unroll
    for (int nt = 0; nt < 4; ++nt) {
      bf16x16 b = ldB_f32(W + (size_t)(ng * 64 + nt * 16) * 512 + kc, 512, lane);
      acc[nt] = wmma_bf16(a, b, acc[nt]);
    }
  }
  if (!transposed) {
#pragma unroll
    for (int nt = 0; nt < 4; ++nt)
      store_tile_bf16(acc[nt], out + (size_t)mt * 16 * 4096 + ng * 64 + nt * 16,
                      4096, lds + widx * 256, lane);
  } else {
    const int sg = mt * 16;
    const int bb = sg >> 10, s0 = (sg & 1023) + ((lane >> 4) << 3);
#pragma unroll
    for (int nt = 0; nt < 4; ++nt) {
      const int col = ng * 64 + nt * 16 + (lane & 15);
      const int h = col >> 9, dk = col & 511;
      bf16x8 vv;
#pragma unroll
      for (int v = 0; v < 8; ++v) vv[v] = (__bf16)acc[nt][v];
      *(bf16x8*)(out + (((size_t)(bb * 8 + h) * 512 + dk) * 1024 + s0)) = vv;
    }
  }
}

// ---------------- Flash attention ----------------
// Block = 4 waves, one (b,h) and 64 query rows; each wave owns 16 rows x DK=512.
// LDS (dynamic, 196KB): Qs[64][512] | Ks[2][32][512] | Vs[2][512][32] | P[4][512]
#define QS_OFF   0
#define KS_OFF   32768           // elements
#define KS_SZ    16384
#define VS_OFF   65536
#define VS_SZ    16384
#define PP_OFF   98304
#define ATTN_LDS_BYTES ((size_t)(98304 + 4 * 512) * 2)

extern __shared__ __bf16 smem_dyn[];

__global__ __launch_bounds__(128) void attn_kernel(const __bf16* __restrict__ qp,
                                                   const __bf16* __restrict__ kp,
                                                   const __bf16* __restrict__ vt,
                                                   __bf16* __restrict__ obuf) {
  const int lane = threadIdx.x & 31, widx = threadIdx.x >> 5;
  const int t = threadIdx.x;
  const int qg = blockIdx.x & 15, h = (blockIdx.x >> 4) & 7, b = blockIdx.x >> 7;

  const __bf16* Qg = qp + ((size_t)(b * 1024 + qg * 64)) * 4096 + h * 512;
  const __bf16* Kg = kp + ((size_t)(b * 1024)) * 4096 + h * 512;
  const __bf16* Vg = vt + ((size_t)(b * 8 + h)) * 512 * 1024;   // vt[dk][s]

  __bf16* Qs = smem_dyn + QS_OFF;
  __bf16* pl = smem_dyn + PP_OFF + widx * 512;
  const float scale = 0.04419417382415922f;  // 1/sqrt(512)

  // ---- stage Q (64x512) : each thread copies half a row (512B = 32 x 16B) ----
  {
    const int r = t >> 1, c0 = (t & 1) * 256;
    const __bf16* g = Qg + (size_t)r * 4096 + c0;
    const unsigned l = lds_off(Qs + r * 512 + c0);
#pragma unroll
    for (int j = 0; j < 32; ++j) async_cp16(l + j * 16, g + j * 8);
  }
  // ---- prefetch K/V block 0 into buffer 0 ----
  {
    const int r = t >> 2, c0 = (t & 3) * 128;          // K: 32 rows x 512
    const __bf16* g = Kg + (size_t)r * 4096 + c0;
    const unsigned l = lds_off(smem_dyn + KS_OFF + r * 512 + c0);
#pragma unroll
    for (int j = 0; j < 16; ++j) async_cp16(l + j * 16, g + j * 8);
  }
  {
#pragma unroll
    for (int rr = 0; rr < 4; ++rr) {                   // V: 512 rows x 32
      const int row = t * 4 + rr;
      const __bf16* g = Vg + (size_t)row * 1024;
      const unsigned l = lds_off(smem_dyn + VS_OFF + row * 32);
#pragma unroll
      for (int j = 0; j < 4; ++j) async_cp16(l + j * 16, g + j * 8);
    }
  }
  asm volatile("s_wait_asynccnt 0" ::: "memory");
  __syncthreads();

  f32x8 o[32];
#pragma unroll
  for (int nt = 0; nt < 32; ++nt) o[nt] = (f32x8){0.f,0.f,0.f,0.f,0.f,0.f,0.f,0.f};
  float m_i[8], l_i[8];
#pragma unroll
  for (int v = 0; v < 8; ++v) { m_i[v] = -1e30f; l_i[v] = 0.f; }

  int cur = 0;
  for (int kb = 0; kb < 1024; kb += 32) {
    // ---- issue async copies for next K/V block into the other buffer ----
    if (kb + 32 < 1024) {
      const int nxt = cur ^ 1;
      {
        const int r = t >> 2, c0 = (t & 3) * 128;
        const __bf16* g = Kg + (size_t)(kb + 32 + r) * 4096 + c0;
        const unsigned l = lds_off(smem_dyn + KS_OFF + nxt * KS_SZ + r * 512 + c0);
#pragma unroll
        for (int j = 0; j < 16; ++j) async_cp16(l + j * 16, g + j * 8);
      }
#pragma unroll
      for (int rr = 0; rr < 4; ++rr) {
        const int row = t * 4 + rr;
        const __bf16* g = Vg + (size_t)row * 1024 + (kb + 32);
        const unsigned l = lds_off(smem_dyn + VS_OFF + nxt * VS_SZ + row * 32);
#pragma unroll
        for (int j = 0; j < 4; ++j) async_cp16(l + j * 16, g + j * 8);
      }
    }

    const __bf16* Ks = smem_dyn + KS_OFF + cur * KS_SZ;
    const __bf16* Vs = smem_dyn + VS_OFF + cur * VS_SZ;
    const __bf16* Qw = Qs + widx * 16 * 512;

    // ---- S = Q K^T (16 queries x 32 keys), all operands from LDS ----
    f32x8 s0 = {}, s1 = {};
#pragma unroll
    for (int kc = 0; kc < 512; kc += 32) {
      bf16x16 a = ldA_bf16(Qw + kc, 512, lane);
      s0 = wmma_bf16(a, ldB_bf16(Ks + kc, 512, lane), s0);
      s1 = wmma_bf16(a, ldB_bf16(Ks + 16 * 512 + kc, 512, lane), s1);
    }

    // ---- streaming softmax stats; row m = v + 8*(lane>>4) ----
    float alpha[8];
#pragma unroll
    for (int v = 0; v < 8; ++v) {
      float a0 = s0[v] * scale, a1 = s1[v] * scale;
      float tt = fmaxf(a0, a1);
      tt = fmaxf(tt, __shfl_xor(tt, 1, 32));
      tt = fmaxf(tt, __shfl_xor(tt, 2, 32));
      tt = fmaxf(tt, __shfl_xor(tt, 4, 32));
      tt = fmaxf(tt, __shfl_xor(tt, 8, 32));
      const float mn = fmaxf(m_i[v], tt);
      alpha[v] = __expf(m_i[v] - mn);
      a0 = __expf(a0 - mn); a1 = __expf(a1 - mn);
      float r = a0 + a1;
      r += __shfl_xor(r, 1, 32);
      r += __shfl_xor(r, 2, 32);
      r += __shfl_xor(r, 4, 32);
      r += __shfl_xor(r, 8, 32);
      l_i[v] = l_i[v] * alpha[v] + r;
      m_i[v] = mn;
      s0[v] = a0; s1[v] = a1;
    }
#pragma unroll
    for (int nt = 0; nt < 32; ++nt)
#pragma unroll
      for (int v = 0; v < 8; ++v) o[nt][v] *= alpha[v];

    // ---- P (C layout) -> LDS row-major -> A operand layout ----
    {
      const int n = lane & 15, mh = (lane >> 4) << 3;
#pragma unroll
      for (int v = 0; v < 8; ++v) {
        pl[(mh + v) * 32 + n]      = (__bf16)s0[v];
        pl[(mh + v) * 32 + 16 + n] = (__bf16)s1[v];
      }
    }
    asm volatile("s_wait_dscnt 0" ::: "memory");
    bf16x16 pa;
    {
      const int r = lane & 15, kh = (lane >> 4) << 3;
      bf16x8 plo = *(const bf16x8*)(pl + r * 32 + kh);
      bf16x8 phi = *(const bf16x8*)(pl + r * 32 + kh + 16);
      asm volatile("s_wait_dscnt 0" ::: "memory");
      pa = __builtin_shufflevector(plo, phi, 0,1,2,3,4,5,6,7,8,9,10,11,12,13,14,15);
    }

    // ---- O += P @ V_block ; B columns are dk -> rows of Vs ----
#pragma unroll
    for (int nt = 0; nt < 32; ++nt) {
      bf16x16 bv = ldB_bf16(Vs + nt * 16 * 32, 32, lane);
      o[nt] = wmma_bf16(pa, bv, o[nt]);
    }

    asm volatile("s_wait_asynccnt 0" ::: "memory");  // next buffer landed
    __syncthreads();                                 // all waves done with cur
    cur ^= 1;
  }

  float inv[8];
#pragma unroll
  for (int v = 0; v < 8; ++v) inv[v] = 1.f / l_i[v];
#pragma unroll
  for (int nt = 0; nt < 32; ++nt)
#pragma unroll
    for (int v = 0; v < 8; ++v) o[nt][v] *= inv[v];

  __bf16* outrow = obuf + ((size_t)(b * 1024 + (qg * 4 + widx) * 16)) * 4096 + h * 512;
#pragma unroll
  for (int nt = 0; nt < 32; ++nt)
    store_tile_bf16(o[nt], outrow + nt * 16, 4096, pl, lane);
}

// ---------------- FC: obuf (8192x4096 bf16) @ w_fc^T (512x4096 f32) -> fcout f32 ----------
__global__ __launch_bounds__(256) void fc_kernel(const __bf16* __restrict__ ob,
                                                 const float* __restrict__ wfc,
                                                 float* __restrict__ fcout) {
  const int lane = threadIdx.x & 31, widx = threadIdx.x >> 5;
  const int wg = blockIdx.x * 8 + widx;
  const int mt = wg >> 3, ng = wg & 7;       // 512 x 8
  const __bf16* At = ob + (size_t)mt * 16 * 4096;
  f32x8 acc[4] = {};
  for (int kc = 0; kc < 4096; kc += 32) {
    bf16x16 a = ldA_bf16(At + kc, 4096, lane);
#pragma unroll
    for (int nt = 0; nt < 4; ++nt) {
      bf16x16 bb = ldB_f32(wfc + (size_t)(ng * 64 + nt * 16) * 4096 + kc, 4096, lane);
      acc[nt] = wmma_bf16(a, bb, acc[nt]);
    }
  }
  const int n = lane & 15, mh = (lane >> 4) << 3;
#pragma unroll
  for (int nt = 0; nt < 4; ++nt) {
    const int col = ng * 64 + nt * 16 + n;
#pragma unroll
    for (int v = 0; v < 8; ++v)
      fcout[(size_t)(mt * 16 + mh + v) * 512 + col] = acc[nt][v];
  }
}

// ---------------- Residual + LayerNorm (one wave per row of 512) ----------------
__global__ __launch_bounds__(128) void ln_kernel(const float* __restrict__ x,
                                                 const float* __restrict__ resid,
                                                 const float* __restrict__ gamma,
                                                 const float* __restrict__ beta,
                                                 float* __restrict__ out) {
  const int lane = threadIdx.x & 31, widx = threadIdx.x >> 5;
  const int row = blockIdx.x * 4 + widx;
  const float* xr = x + (size_t)row * 512;
  const float* rr = resid + (size_t)row * 512;
  float vbuf[16];
  float s = 0.f, s2 = 0.f;
#pragma unroll
  for (int j = 0; j < 16; ++j) {
    const float t = xr[lane + j * 32] + rr[lane + j * 32];
    vbuf[j] = t; s += t; s2 += t * t;
  }
#pragma unroll
  for (int m = 1; m <= 16; m <<= 1) {
    s  += __shfl_xor(s,  m, 32);
    s2 += __shfl_xor(s2, m, 32);
  }
  const float mu  = s * (1.f / 512.f);
  const float var = s2 * (1.f / 512.f) - mu * mu;
  const float rs  = rsqrtf(var + 1e-6f);
  float* orow = out + (size_t)row * 512;
#pragma unroll
  for (int j = 0; j < 16; ++j) {
    const int c = lane + j * 32;
    orow[c] = (vbuf[j] - mu) * rs * gamma[c] + beta[c];
  }
}

extern "C" void kernel_launch(void* const* d_in, const int* in_sizes, int n_in,
                              void* d_out, int out_size, void* d_ws, size_t ws_size,
                              hipStream_t stream) {
  (void)in_sizes; (void)n_in; (void)out_size; (void)ws_size;
  const float* q     = (const float*)d_in[0];
  const float* k     = (const float*)d_in[1];
  const float* v     = (const float*)d_in[2];
  const float* w_q   = (const float*)d_in[3];
  const float* w_k   = (const float*)d_in[4];
  const float* w_v   = (const float*)d_in[5];
  const float* w_fc  = (const float*)d_in[6];
  const float* gamma = (const float*)d_in[7];
  const float* beta  = (const float*)d_in[8];
  float* out = (float*)d_out;

  // workspace: qp/kp/vt/obuf 64MB each (bf16), fcout 16MB (f32) -> 272MB total
  char* ws = (char*)d_ws;
  __bf16* qp    = (__bf16*)(ws);
  __bf16* kp    = (__bf16*)(ws + ((size_t)64  << 20));
  __bf16* vt    = (__bf16*)(ws + ((size_t)128 << 20));
  __bf16* obuf  = (__bf16*)(ws + ((size_t)192 << 20));
  float*  fcout = (float*) (ws + ((size_t)256 << 20));

  static bool attr_set = false;
  if (!attr_set) {
    hipFuncSetAttribute((const void*)attn_kernel,
                        hipFuncAttributeMaxDynamicSharedMemorySize,
                        (int)ATTN_LDS_BYTES);
    attr_set = true;
  }

  proj_kernel<<<4096, 256, 0, stream>>>(q, w_q, qp, 0);
  proj_kernel<<<4096, 256, 0, stream>>>(k, w_k, kp, 0);
  proj_kernel<<<4096, 256, 0, stream>>>(v, w_v, vt, 1);
  attn_kernel<<<1024, 128, ATTN_LDS_BYTES, stream>>>(qp, kp, vt, obuf);
  fc_kernel<<<512, 256, 0, stream>>>(obuf, w_fc, fcout);
  ln_kernel<<<2048, 128, 0, stream>>>(fcout, q, gamma, beta, out);
}
